// protein_conv_17626545783636
// MI455X (gfx1250) — matrix operand order repaired and checked by reading, not compiled, not run on hardware
//
#include <hip/hip_runtime.h>
#include <hip/hip_bf16.h>
#include <stdint.h>
#include <stddef.h>

// ---------------------------------------------------------------------------
// Problem constants (from the reference)
// ---------------------------------------------------------------------------
#define NN    50000   // nodes
#define EE    800000  // edges
#define HH    128     // hidden channels
#define PCF   1280    // pc feature length
#define POSF  16
#define EDGEF 8

typedef __attribute__((ext_vector_type(16))) __bf16 v16bf;
typedef __attribute__((ext_vector_type(8)))  float  v8f;
typedef __attribute__((ext_vector_type(4)))  unsigned int u32x4;
typedef __attribute__((ext_vector_type(8)))  int          i32x8;
typedef __attribute__((ext_vector_type(4)))  int          i32x4;

// ---------------------------------------------------------------------------
// Generic WMMA GEMM:  C[:, col0:col0+128] = [relu]( (A (+A2)) * rowscale @ W + b  (+ Cadd) )
//   A : M x K row-major (lda), W : K x 128 row-major, K multiple of 32.
//   One wave computes a 16-row strip x all 128 output columns (8 f32 accums).
//   W chunks (32 K-rows) are staged into LDS by the whole block, pre-converted
//   to bf16 and pre-swizzled into per-(tile,lane) B fragments, double-buffered:
//   compute-side B fragment = 2x ds_load_b128 instead of 16 strided VMEM loads.
// ---------------------------------------------------------------------------
__global__ void __launch_bounds__(256)
wmma_gemm128(const float* __restrict__ A,
             const float* __restrict__ A2,        // optional elementwise add to A or nullptr
             const float* __restrict__ rowscale,  // optional per-row scale of A or nullptr
             int lda,
             const float* __restrict__ W,         // K x 128
             const float* __restrict__ bias,      // 128
             const float* __restrict__ Cadd,      // optional residual [M x 128], ld = ldadd
             int ldadd,
             float* __restrict__ C, int ldc, int col0,
             int M, int K, int do_relu)
{
    // [parity][ (tile*32 + lane) * 32B ] : 16 bf16 per fragment slot, 2 x 8 KB
    __shared__ __align__(32) uint4 sB4[2][512];

    const int wid   = threadIdx.x >> 5;
    const int lane  = threadIdx.x & 31;
    const int strip = blockIdx.x * 8 + wid;   // 16-row strip per wave
    const int row0  = strip * 16;
    const bool active = (row0 < M);

    const int lh   = lane >> 4;     // lane half: 0 or 1
    const int rrow = lane & 15;

    const int arow = active ? (row0 + rrow) : rrow;   // clamp inactive waves to row 0..15
    const float rs = rowscale ? rowscale[arow] : 1.0f;
    const float* __restrict__ Ap  = A  + (size_t)arow * lda;
    const float* __restrict__ A2p = A2 ? (A2 + (size_t)arow * lda) : nullptr;

    // staging role of this thread: fragment slot (tile, lane) -- 256 slots, 1:1
    const int st   = threadIdx.x >> 5;          // tile 0..7
    const int sl   = threadIdx.x & 31;          // lane 0..31
    const int scol = st * 16 + (sl & 15);
    const int skh  = 8 * (sl >> 4);
    uint4* const sslot = &sB4[0][0] + ((st << 5) + sl) * 2;   // parity adds +512

    // stage one 32-K-row chunk of W into LDS buffer p (bf16, fragment-swizzled)
    auto stage = [&](int k0, int p) {
        const float* wp = W + (size_t)(k0 + skh) * HH + scol;  // 16 rows, stride HH
        v16bf vb;
#pragma unroll
        for (int i = 0; i < 16; i++) vb[i] = (__bf16)wp[(size_t)(i + 8 * (i >> 3)) * HH];
        *(v16bf*)(sslot + p * 512) = vb;
    };

    v8f acc[8] = {};   // 8 column tiles of 16x16 f32

    stage(0, 0);
    __syncthreads();

    for (int k0 = 0; k0 < K; k0 += 32) {
        const int p = (k0 >> 5) & 1;
        if (k0 + 32 < K) {
            __builtin_prefetch(Ap + k0 + 32, 0, 0);  // global_prefetch_b8
            stage(k0 + 32, p ^ 1);
        }

        // ---- A fragment (16x32 bf16): elem i  -> K = k0 + 8*lh + i
        //                               elem i+8-> K = k0 + 16 + 8*lh + i
        float4 f0 = *(const float4*)(Ap + k0 + 8 * lh);
        float4 f1 = *(const float4*)(Ap + k0 + 8 * lh + 4);
        float4 f2 = *(const float4*)(Ap + k0 + 16 + 8 * lh);
        float4 f3 = *(const float4*)(Ap + k0 + 16 + 8 * lh + 4);
        if (A2p) {
            const float4 g0 = *(const float4*)(A2p + k0 + 8 * lh);
            const float4 g1 = *(const float4*)(A2p + k0 + 8 * lh + 4);
            const float4 g2 = *(const float4*)(A2p + k0 + 16 + 8 * lh);
            const float4 g3 = *(const float4*)(A2p + k0 + 16 + 8 * lh + 4);
            f0.x += g0.x; f0.y += g0.y; f0.z += g0.z; f0.w += g0.w;
            f1.x += g1.x; f1.y += g1.y; f1.z += g1.z; f1.w += g1.w;
            f2.x += g2.x; f2.y += g2.y; f2.z += g2.z; f2.w += g2.w;
            f3.x += g3.x; f3.y += g3.y; f3.z += g3.z; f3.w += g3.w;
        }
        const float av[16] = { f0.x, f0.y, f0.z, f0.w, f1.x, f1.y, f1.z, f1.w,
                               f2.x, f2.y, f2.z, f2.w, f3.x, f3.y, f3.z, f3.w };
        v16bf a;
#pragma unroll
        for (int i = 0; i < 16; i++) a[i] = (__bf16)(av[i] * rs);

        const uint4* bbase = &sB4[p][lane * 2];   // + t*64 B fragment stride
#pragma unroll
        for (int t = 0; t < 8; t++) {
            const v16bf b = *(const v16bf*)(bbase + t * 64);   // 2x ds_load_b128
            acc[t] = __builtin_amdgcn_wmma_f32_16x16x32_bf16(
                         false, a, false, b, (short)0, acc[t], false, false);
        }
        __syncthreads();
    }

    if (!active) return;

    // ---- epilogue: C/D layout -> M = row0 + r + 8*lh, N = t*16 + rrow
#pragma unroll
    for (int t = 0; t < 8; t++) {
        const int n = t * 16 + rrow;
        const float bv = bias[n];
#pragma unroll
        for (int r = 0; r < 8; r++) {
            const int m = row0 + 8 * lh + r;
            float v = acc[t][r] + bv;
            if (Cadd) v += Cadd[(size_t)m * ldadd + n];
            if (do_relu) v = fmaxf(v, 0.0f);
            C[(size_t)m * ldc + col0 + n] = v;
        }
    }
}

// ---------------------------------------------------------------------------
// Fused edge conv message+scatter:
//   ew = feat[e] @ Wf + bf   (Kf x 128 weight staged into LDS by the TDM)
//   atomicAdd(agg[dst[e]], ew * x[src[e]])
// One wave per edge; each lane owns 4 channels. Gather/scatter bound; the
// Kf<=16 mat-vec hides under the x[src] gather; atomics land in the 192MB L2.
// ---------------------------------------------------------------------------
__global__ void __launch_bounds__(256)
edge_conv_fused(const float* __restrict__ feat, int Kf,
                const float* __restrict__ Wf,   // Kf x 128
                const float* __restrict__ bf,   // 128
                const float* __restrict__ x,    // N x 128 (x1)
                const int*   __restrict__ src,
                const int*   __restrict__ dst,
                float*       __restrict__ agg,  // N x 128, pre-zeroed
                int nE)
{
    __shared__ float sW[POSF * HH];   // max 16*128 f32
    __shared__ float sB[HH];

    // ---- Tensor Data Mover: DMA the Kf x 128 f32 weight tile into LDS.
    // D# group0/group1 packed per CDNA5 ISA 8.3/8.4 (count=1, data_size=4B,
    // type=2, tensor 128 x Kf, tile 128 x Kf, dim0_stride=128).
    if (threadIdx.x == 0) {
        const uint64_t ga  = (uint64_t)(uintptr_t)Wf;
        const uint32_t lds = (uint32_t)(uintptr_t)(void*)&sW[0];  // low 32b = LDS offset
        u32x4 g0;
        g0[0] = 1u;                                   // count=1, user mode
        g0[1] = lds;                                  // lds_addr
        g0[2] = (uint32_t)ga;                         // global_addr[31:0]
        g0[3] = (uint32_t)((ga >> 32) & 0x1FFFFFFu)   // global_addr[56:32]
              | (2u << 30);                           // type=2 ("image")
        i32x8 g1;
        g1[0] = 2 << 16;                              // data_size=2 -> 4 bytes
        g1[1] = HH << 16;                             // tensor_dim0[15:0]=128 @ bits 63:48
        g1[2] = Kf << 16;                             // tensor_dim1[15:0]=Kf @ bits 111:80
        g1[3] = HH << 16;                             // tile_dim0=128 @ bits 127:112
        g1[4] = Kf;                                   // tile_dim1=Kf, tile_dim2=0
        g1[5] = HH;                                   // tensor_dim0_stride[31:0]=128
        g1[6] = 0;
        g1[7] = 0;
        const i32x4 z4 = {0, 0, 0, 0};                // groups 2/3 unused (2D tensor)
        const i32x8 z8 = {0, 0, 0, 0, 0, 0, 0, 0};    // extra group (clang-23 6-arg form)
        __builtin_amdgcn_tensor_load_to_lds(g0, g1, z4, z4, z8, 0);
        __builtin_amdgcn_s_wait_tensorcnt(0);
    }
    for (int i = threadIdx.x; i < HH; i += blockDim.x) sB[i] = bf[i];
    __syncthreads();

    const int wid  = threadIdx.x >> 5;
    const int lane = threadIdx.x & 31;
    const int e    = blockIdx.x * 8 + wid;
    if (e >= nE) return;

    const int s = src[e];
    const int d = dst[e];
    const int j = lane * 4;

    float4 ew = make_float4(sB[j], sB[j + 1], sB[j + 2], sB[j + 3]);
    const float* __restrict__ fe = feat + (size_t)e * Kf;
    for (int k = 0; k < Kf; k++) {
        const float fv = fe[k];
        const float* wr = sW + k * HH + j;
        ew.x += fv * wr[0];
        ew.y += fv * wr[1];
        ew.z += fv * wr[2];
        ew.w += fv * wr[3];
    }

    const float4 xv = *(const float4*)(x + (size_t)s * HH + j);
    float* ap = agg + (size_t)d * HH + j;
    atomicAdd(ap + 0, ew.x * xv.x);
    atomicAdd(ap + 1, ew.y * xv.y);
    atomicAdd(ap + 2, ew.z * xv.z);
    atomicAdd(ap + 3, ew.w * xv.w);
}

// ---------------------------------------------------------------------------
// Helpers
// ---------------------------------------------------------------------------
__global__ void fill_f32(float* __restrict__ p, size_t n, float v) {
    size_t i = (size_t)blockIdx.x * blockDim.x + threadIdx.x;
    if (i < n) p[i] = v;
}

__global__ void degree_accum(const int* __restrict__ dst, float* __restrict__ deg, int nE) {
    int e = blockIdx.x * blockDim.x + threadIdx.x;
    if (e < nE) atomicAdd(&deg[dst[e]], 1.0f);
}

__global__ void invert_deg(float* __restrict__ deg, int n) {
    int i = blockIdx.x * blockDim.x + threadIdx.x;
    if (i < n) deg[i] = 1.0f / fmaxf(deg[i], 1.0f);
}

// out[n, 0:128] = pc[n], out[n, 128:256] = enc[n]
__global__ void concat_out(const float* __restrict__ pc, const float* __restrict__ enc,
                           float* __restrict__ out, int n) {
    size_t i = (size_t)blockIdx.x * blockDim.x + threadIdx.x;
    if (i >= (size_t)n * 256) return;
    const int row = (int)(i >> 8);
    const int c   = (int)(i & 255);
    out[i] = (c < HH) ? pc[(size_t)row * HH + c] : enc[(size_t)row * HH + (c - HH)];
}

// ---------------------------------------------------------------------------
// Launch
// ---------------------------------------------------------------------------
extern "C" void kernel_launch(void* const* d_in, const int* in_sizes, int n_in,
                              void* d_out, int out_size, void* d_ws, size_t ws_size,
                              hipStream_t stream)
{
    (void)in_sizes; (void)n_in; (void)out_size; (void)ws_size;

    const float* x_in = (const float*)d_in[0];     // [N, 1280]
    const float* pos  = (const float*)d_in[1];     // [E, 16]
    const float* efe  = (const float*)d_in[2];     // [E, 8]
    const int*   eidx = (const int*)  d_in[3];     // [2, E]
    const int*   src  = eidx;
    const int*   dst  = eidx + EE;

    auto P = [&](int i) { return (const float*)d_in[i]; };
    const float *Wpc = P(4), *bpc = P(5);
    const float *Wx  = P(6), *bx  = P(7);

    // ---- workspace layout (all f32): 50000*(6*128 + 256 + 1)*4 = 205 MB
    float* pc   = (float*)d_ws;
    float* enc  = pc   + (size_t)NN * HH;
    float* x1   = enc  + (size_t)NN * HH;
    float* x2   = x1   + (size_t)NN * HH;
    float* tbuf = x2   + (size_t)NN * HH;
    float* agg  = tbuf + (size_t)NN * HH;
    float* hcat = agg  + (size_t)NN * HH;      // N x 256
    float* deg  = hcat + (size_t)NN * 2 * HH;  // N

    const dim3 blk(256);
    const int gemmGrid = ((NN / 16) + 7) / 8;                      // 391 blocks
    const int fillNH   = (int)(((size_t)NN * HH + 255) / 256);
    const int edgeGrid = (EE + 7) / 8;

    // ---- in-degree -> inv_deg (shared by both blocks)
    fill_f32<<<(NN + 255) / 256, blk, 0, stream>>>(deg, (size_t)NN, 0.0f);
    degree_accum<<<(EE + 255) / 256, blk, 0, stream>>>(dst, deg, EE);
    invert_deg<<<(NN + 255) / 256, blk, 0, stream>>>(deg, NN);

    // ---- pc = relu(x @ Wpc + b);  enc = relu(pc @ Wx + b)
    wmma_gemm128<<<gemmGrid, blk, 0, stream>>>(x_in, nullptr, nullptr, PCF, Wpc, bpc,
                                               nullptr, 0, pc, HH, 0, NN, PCF, 1);
    wmma_gemm128<<<gemmGrid, blk, 0, stream>>>(pc, nullptr, nullptr, HH, Wx, bx,
                                               nullptr, 0, enc, HH, 0, NN, HH, 1);

    for (int b = 0; b < 2; b++) {
        const int o = 8 + 26 * b;
        const float *Wx1 = P(o+0),  *bx1 = P(o+1),  *Wx2 = P(o+2),  *bx2 = P(o+3);
        const float *Wp1 = P(o+4),  *bp1 = P(o+5),  *Wp2 = P(o+6),  *bp2 = P(o+7);
        const float *We1 = P(o+8),  *be1 = P(o+9),  *We2 = P(o+10), *be2 = P(o+11);
        const float *W0l = P(o+12), *b0l = P(o+13), *W0r = P(o+14), *b0r = P(o+15);
        const float *W1l = P(o+16), *b1l = P(o+17), *W1r = P(o+18), *b1r = P(o+19);
        const float *Wc0 = P(o+20), *bc0 = P(o+21), *Wc1 = P(o+22), *bc1 = P(o+23);
        const float *Wfn = P(o+24), *bfn = P(o+25);

        // x1 = relu(lin_x1(enc)); x2 = relu(lin_x2(enc))
        wmma_gemm128<<<gemmGrid, blk, 0, stream>>>(enc, nullptr, nullptr, HH, Wx1, bx1,
                                                   nullptr, 0, x1, HH, 0, NN, HH, 1);
        wmma_gemm128<<<gemmGrid, blk, 0, stream>>>(enc, nullptr, nullptr, HH, Wx2, bx2,
                                                   nullptr, 0, x2, HH, 0, NN, HH, 1);

        // conv0: agg = scatter_sum(lin_pos1(pos) * x1[src]); t = (agg*inv)@W0l+b + x1@W0r+b
        fill_f32<<<fillNH, blk, 0, stream>>>(agg, (size_t)NN * HH, 0.0f);
        edge_conv_fused<<<edgeGrid, blk, 0, stream>>>(pos, POSF, Wp1, bp1, x1, src, dst, agg, EE);
        wmma_gemm128<<<gemmGrid, blk, 0, stream>>>(agg, nullptr, deg, HH, W0l, b0l,
                                                   nullptr, 0, tbuf, HH, 0, NN, HH, 0);
        wmma_gemm128<<<gemmGrid, blk, 0, stream>>>(x1, nullptr, nullptr, HH, W0r, b0r,
                                                   tbuf, HH, tbuf, HH, 0, NN, HH, 0);
        // h0 = relu(lin_pos2(t)) -> hcat[:, 0:128]
        wmma_gemm128<<<gemmGrid, blk, 0, stream>>>(tbuf, nullptr, nullptr, HH, Wp2, bp2,
                                                   nullptr, 0, hcat, 2 * HH, 0, NN, HH, 1);

        // conv1 with edge features
        fill_f32<<<fillNH, blk, 0, stream>>>(agg, (size_t)NN * HH, 0.0f);
        edge_conv_fused<<<edgeGrid, blk, 0, stream>>>(efe, EDGEF, We1, be1, x1, src, dst, agg, EE);
        wmma_gemm128<<<gemmGrid, blk, 0, stream>>>(agg, nullptr, deg, HH, W1l, b1l,
                                                   nullptr, 0, tbuf, HH, 0, NN, HH, 0);
        wmma_gemm128<<<gemmGrid, blk, 0, stream>>>(x1, nullptr, nullptr, HH, W1r, b1r,
                                                   tbuf, HH, tbuf, HH, 0, NN, HH, 0);
        // h1 = relu(lin_edge2(t)) -> hcat[:, 128:256]
        wmma_gemm128<<<gemmGrid, blk, 0, stream>>>(tbuf, nullptr, nullptr, HH, We2, be2,
                                                   nullptr, 0, hcat, 2 * HH, HH, NN, HH, 1);

        // hc = relu(cat0(hcat)) -> x1 (dead);  hc = relu(cat1(hc)) -> tbuf
        wmma_gemm128<<<gemmGrid, blk, 0, stream>>>(hcat, nullptr, nullptr, 2 * HH, Wc0, bc0,
                                                   nullptr, 0, x1, HH, 0, NN, 2 * HH, 1);
        wmma_gemm128<<<gemmGrid, blk, 0, stream>>>(x1, nullptr, nullptr, HH, Wc1, bc1,
                                                   nullptr, 0, tbuf, HH, 0, NN, HH, 1);

        // enc = final(hc + x2) + enc   (A2 fuses +x2, Cadd fuses +enc; in-place safe:
        // each output element is read once then written by the same lane)
        wmma_gemm128<<<gemmGrid, blk, 0, stream>>>(tbuf, x2, nullptr, HH, Wfn, bfn,
                                                   enc, HH, enc, HH, 0, NN, HH, 0);
    }

    // out = concat([pc, enc], axis=1)
    concat_out<<<(int)(((size_t)NN * 256 + 255) / 256), blk, 0, stream>>>(pc, enc, (float*)d_out, NN);
}
